// WindowAttention_66383014527689
// MI455X (gfx1250) — matrix operand order repaired and checked by reading, not compiled, not run on hardware
//
#include <hip/hip_runtime.h>
#include <hip/hip_bf16.h>
#include <math.h>

// ---------------------------------------------------------------------------
// Window attention, fused, bf16 WMMA (gfx1250 / MI455X).
//   DIM=512 NH=16 CH=32 L=343 N=64  -> rows = 21952
// Pipeline:
//   cvt f32->bf16 (x, W_qkv, W_proj)
//   qkv GEMM  (bf16 wmma 16x16x32, double-buffered LDS, async Global->LDS)
//   flash attention (scores wmma K=CH=32, online softmax, PV wmma)
//   proj GEMM -> f32 out + bias
// ---------------------------------------------------------------------------

#define DIMC   512
#define NHC    16
#define CHC    32
#define LC     343
#define TABLEC 2197
#define NWIN   64
#define ROWS   (NWIN * LC)          // 21952
#define QSCALE 0.17677669529663687f // CH^-0.5
#define LDA    40                   // LDS row stride in bf16 elems (80B, 16B-aligned)
#define BUFE   (64 * LDA)           // elems per LDS tile buffer

typedef __attribute__((ext_vector_type(16))) __bf16 v16bf;
typedef __attribute__((ext_vector_type(8)))  float  v8f;

// ---------------- async Global->LDS copy (gfx1250), with safe fallback -----

#if defined(__HIP_DEVICE_COMPILE__) && __has_builtin(__builtin_amdgcn_global_load_async_to_lds_b128)
#define USE_ASYNC_LDS 1
#else
#define USE_ASYNC_LDS 0
#endif

#if USE_ASYNC_LDS
// builtin signature: void(int4 AS1* src, int4 AS3* dst, imm int off, imm int cpol)
typedef int v4i_gcc __attribute__((vector_size(16)));
typedef __attribute__((address_space(1))) v4i_gcc as1_v4i;
typedef __attribute__((address_space(3))) v4i_gcc as3_v4i;
// copy 16 bytes memory -> LDS per lane, tracked by ASYNCcnt
__device__ __forceinline__ void cp16_async(void* lds, const void* g) {
    __builtin_amdgcn_global_load_async_to_lds_b128((as1_v4i*)g, (as3_v4i*)lds, 0, 0);
}
__device__ __forceinline__ void wait_async0() {
#if __has_builtin(__builtin_amdgcn_s_wait_asynccnt)
    __builtin_amdgcn_s_wait_asynccnt(0);
#else
    asm volatile("s_wait_asynccnt 0x0" ::: "memory");
#endif
}
#endif

// ---------------- fragment loaders (per CDNA5 ISA 7.12.2 layouts) ----------

// A 16x32 bf16, source row-major [16][stride]
__device__ inline v16bf ld_frag_a(const __bf16* base, int stride) {
    const int lane = threadIdx.x & 31;
    const int half = lane >> 4;
    const int r    = lane & 15;
    const __bf16* p = base + r * stride;
    v16bf a;
    __builtin_memcpy(&a, p + half * 8, 16);                                    // K = half*8 .. +7
    __builtin_memcpy(reinterpret_cast<char*>(&a) + 16, p + 16 + half * 8, 16); // K = 16+half*8 .. +7
    return a;
}

// B 32x16 bf16, source row-major [N=16][K=32] (i.e. B[k][n] = src[n][k])
__device__ inline v16bf ld_frag_b(const __bf16* base, int stride) {
    const int lane = threadIdx.x & 31;
    const int half = lane >> 4;
    const int n    = lane & 15;
    v16bf b;
    __builtin_memcpy(&b, base + n * stride + half * 8, 16);                                    // K = half*8 .. +7
    __builtin_memcpy(reinterpret_cast<char*>(&b) + 16, base + n * stride + 16 + half * 8, 16); // K = 16+half*8 .. +7
    return b;
}

__device__ inline v8f wmma_bf16(v16bf a, v16bf b, v8f c) {
    return __builtin_amdgcn_wmma_f32_16x16x32_bf16(false, a, false, b, (short)0, c, false, false);
}

__device__ inline float redmax16(float v) {
#pragma unroll
    for (int i = 1; i < 16; i <<= 1) v = fmaxf(v, __shfl_xor(v, i, 32));
    return v;
}
__device__ inline float redsum16(float v) {
#pragma unroll
    for (int i = 1; i < 16; i <<= 1) v += __shfl_xor(v, i, 32);
    return v;
}

// ---------------- f32 -> bf16 convert --------------------------------------

__global__ void cvt_bf16(const float* __restrict__ in, __bf16* __restrict__ out, int n) {
    int i = blockIdx.x * blockDim.x + threadIdx.x;
    const int stride = gridDim.x * blockDim.x;
    for (; i < n; i += stride) out[i] = (__bf16)in[i];
}

// ---------------- shared GEMM mainloop: C[64x64] tile ----------------------
// A row-major [M][K=512] bf16, B row-major [N][K=512] bf16 (computes A@B^T).
// Block = 128 threads (4 waves); wave w owns M-rows [mb+16w, +16), all 64 N.
// LDS tiles are double-buffered; tile t+1 is fetched while tile t computes.

__device__ inline void gemm_mainloop(const __bf16* __restrict__ A,
                                     const __bf16* __restrict__ B,
                                     int mb, int nb,
                                     __bf16* sA, __bf16* sB, v8f acc[4]) {
    const int tid = threadIdx.x;
    const int w   = tid >> 5;
    const int row = tid >> 1;         // 0..63
    const int seg = (tid & 1) * 16;   // bf16 elems
    const __bf16* gA = A + (size_t)(mb + row) * DIMC + seg;
    const __bf16* gB = B + (size_t)(nb + row) * DIMC + seg;
    __bf16* lA = sA + row * LDA + seg;
    __bf16* lB = sB + row * LDA + seg;

#if USE_ASYNC_LDS
    // prologue: async-fetch tile 0 into buffer 0
    cp16_async(lA, gA);          cp16_async(lA + 8, gA + 8);
    cp16_async(lB, gB);          cp16_async(lB + 8, gB + 8);
    wait_async0();
    __syncthreads();
    for (int kt = 0; kt < DIMC; kt += 32) {
        const int cur = (kt >> 5) & 1;
        const int nxt = cur ^ 1;
        if (kt + 32 < DIMC) {   // async-fetch next tile while computing
            cp16_async(lA + nxt * BUFE,     gA + kt + 32);
            cp16_async(lA + nxt * BUFE + 8, gA + kt + 40);
            cp16_async(lB + nxt * BUFE,     gB + kt + 32);
            cp16_async(lB + nxt * BUFE + 8, gB + kt + 40);
        }
        const __bf16* bA = sA + cur * BUFE + w * 16 * LDA;
        const __bf16* bB = sB + cur * BUFE;
        v16bf a = ld_frag_a(bA, LDA);
#pragma unroll
        for (int t = 0; t < 4; ++t) {
            v16bf b = ld_frag_b(bB + t * 16 * LDA, LDA);
            acc[t] = wmma_bf16(a, b, acc[t]);
        }
        wait_async0();
        __syncthreads();
    }
#else
    // fallback: registers-first staging (loads issue before any wait)
    uint4 ra0, ra1, rb0, rb1;
    {
        const uint4* ga = reinterpret_cast<const uint4*>(gA);
        const uint4* gb = reinterpret_cast<const uint4*>(gB);
        ra0 = ga[0]; ra1 = ga[1]; rb0 = gb[0]; rb1 = gb[1];
    }
    {
        uint4* la = reinterpret_cast<uint4*>(lA);
        uint4* lb = reinterpret_cast<uint4*>(lB);
        la[0] = ra0; la[1] = ra1; lb[0] = rb0; lb[1] = rb1;
    }
    __syncthreads();
    for (int kt = 0; kt < DIMC; kt += 32) {
        const int cur  = (kt >> 5) & 1;
        const int nxt  = cur ^ 1;
        const bool more = (kt + 32 < DIMC);
        if (more) {  // issue next-tile global loads before compute
            const uint4* ga = reinterpret_cast<const uint4*>(gA + kt + 32);
            const uint4* gb = reinterpret_cast<const uint4*>(gB + kt + 32);
            ra0 = ga[0]; ra1 = ga[1]; rb0 = gb[0]; rb1 = gb[1];
        }
        const __bf16* bA = sA + cur * BUFE + w * 16 * LDA;
        const __bf16* bB = sB + cur * BUFE;
        v16bf a = ld_frag_a(bA, LDA);
#pragma unroll
        for (int t = 0; t < 4; ++t) {
            v16bf b = ld_frag_b(bB + t * 16 * LDA, LDA);
            acc[t] = wmma_bf16(a, b, acc[t]);
        }
        if (more) {
            uint4* la = reinterpret_cast<uint4*>(sA + nxt * BUFE + row * LDA + seg);
            uint4* lb = reinterpret_cast<uint4*>(sB + nxt * BUFE + row * LDA + seg);
            la[0] = ra0; la[1] = ra1; lb[0] = rb0; lb[1] = rb1;
        }
        __syncthreads();
    }
#endif
}

// ---------------- kernel 1: QKV projection ---------------------------------
// grid (24, 343): x = N-tile over 1536, y = M-tile over 21952.
// Scatter into q/k/v [n][h][l][c] bf16; q pre-scaled by CH^-0.5.

__global__ void __launch_bounds__(128)
qkv_gemm(const __bf16* __restrict__ xbf, const __bf16* __restrict__ wq,
         __bf16* __restrict__ qb, __bf16* __restrict__ kb, __bf16* __restrict__ vb) {
    __shared__ __bf16 sA[2 * BUFE];
    __shared__ __bf16 sB[2 * BUFE];
    const int mb = blockIdx.y * 64;
    const int nb = blockIdx.x * 64;
    v8f acc[4] = {{0,0,0,0,0,0,0,0},{0,0,0,0,0,0,0,0},{0,0,0,0,0,0,0,0},{0,0,0,0,0,0,0,0}};
    gemm_mainloop(xbf, wq, mb, nb, sA, sB, acc);

    const int lane = threadIdx.x & 31, half = lane >> 4, nl = lane & 15;
    const int w = threadIdx.x >> 5;
#pragma unroll
    for (int t = 0; t < 4; ++t) {
        const int gcol  = nb + t * 16 + nl;
        const int which = gcol >> 9;       // 0=q 1=k 2=v
        const int d     = gcol & 511;
        const int h     = d >> 5;
        const int c     = d & 31;
#pragma unroll
        for (int r = 0; r < 8; ++r) {
            const int grow = mb + w * 16 + r + 8 * half;
            const int nw   = grow / LC;
            const int l    = grow - nw * LC;
            const size_t dst = (((size_t)nw * NHC + h) * LC + l) * CHC + c;
            const float v = acc[t][r];
            if (which == 0)      qb[dst] = (__bf16)(v * QSCALE);
            else if (which == 1) kb[dst] = (__bf16)v;
            else                 vb[dst] = (__bf16)v;
        }
    }
}

// ---------------- kernel 2: fused flash attention --------------------------
// grid (6, NH, NWIN), block 128 (4 waves). Wave w: 16-query tile at
// qbase = 64*blockIdx.x + 16*w. Keys processed in blocks of 32 (11 blocks).

__global__ void __launch_bounds__(128)
attn_kernel(const __bf16* __restrict__ qb, const __bf16* __restrict__ kb,
            const __bf16* __restrict__ vb, const int* __restrict__ rpi,
            const float* __restrict__ btab, const unsigned char* __restrict__ maskp,
            __bf16* __restrict__ abuf) {
    __shared__ __bf16 sK[32 * LDA];        // [key][ch]
    __shared__ __bf16 sV[32 * LDA];        // [ch][key]  (transposed at staging)
    __shared__ __bf16 sP[4 * 16 * LDA];    // per-wave probability tiles [16][32]

    const int tid = threadIdx.x;
    const int w = tid >> 5, lane = tid & 31, half = lane >> 4, nl = lane & 15;
    const int n = blockIdx.z, h = blockIdx.y;
    const int qbase = blockIdx.x * 64 + w * 16;
    const size_t headbase = ((size_t)n * NHC + h) * LC * CHC;
    const int hT = h * TABLEC;
    __bf16* Pw = sP + w * 16 * LDA;
    const float NEG_INF = -__builtin_inff();

    // q A-fragment, resident in registers for the whole loop
    v16bf aq;
    {
        int qr = qbase + nl; if (qr > LC - 1) qr = LC - 1;
        const __bf16* qp = qb + headbase + (size_t)qr * CHC + half * 8;
        __builtin_memcpy(&aq, qp, 16);
        __builtin_memcpy(reinterpret_cast<char*>(&aq) + 16, qp + 16, 16);
    }

    float m[8], s[8];
    v8f o0 = {0,0,0,0,0,0,0,0}, o1 = {0,0,0,0,0,0,0,0};
#pragma unroll
    for (int r = 0; r < 8; ++r) { m[r] = NEG_INF; s[r] = 0.f; }

    for (int kb0 = 0; kb0 < 352; kb0 += 32) {
        // ---- stage K tile [32 keys][32 ch] and transposed V tile [32 ch][32 keys]
        {
            const int key = tid >> 2, cp = (tid & 3) * 8;
            int kr = kb0 + key; if (kr > LC - 1) kr = LC - 1;
            const __bf16* gk = kb + headbase + (size_t)kr * CHC + cp;
#if USE_ASYNC_LDS
            cp16_async(sK + key * LDA + cp, gk);      // DMA, no VGPR round-trip
#else
            const uint4 kreg = *reinterpret_cast<const uint4*>(gk);
#endif
            const int vkey = tid & 31, cbv = (tid >> 5) * 8;
            int vr = kb0 + vkey; if (vr > LC - 1) vr = LC - 1;
            __bf16 tmp[8];
            __builtin_memcpy(tmp, vb + headbase + (size_t)vr * CHC + cbv, 16);
#pragma unroll
            for (int i = 0; i < 8; ++i) sV[(cbv + i) * LDA + vkey] = tmp[i];
#if USE_ASYNC_LDS
            wait_async0();
#else
            *reinterpret_cast<uint4*>(sK + key * LDA + cp) = kreg;
#endif
        }
        __syncthreads();

        // ---- scores: q(16x32) @ k^T -> two 16x16 f32 tiles
        v16bf bk0 = ld_frag_b(sK, LDA);
        v16bf bk1 = ld_frag_b(sK + 16 * LDA, LDA);
        v8f s0 = {0,0,0,0,0,0,0,0}, s1 = {0,0,0,0,0,0,0,0};
        s0 = wmma_bf16(aq, bk0, s0);
        s1 = wmma_bf16(aq, bk1, s1);

        // ---- bias + mask + online softmax (row = r + 8*half, col = nl)
#pragma unroll
        for (int r = 0; r < 8; ++r) {
            int qg = qbase + r + 8 * half;
            int qc = qg > LC - 1 ? LC - 1 : qg;
            const size_t ridx = (size_t)qc * LC;
            const size_t mrow = ((size_t)n * LC + qc) * LC;
            const int k0 = kb0 + nl, k1 = k0 + 16;
            float sc0, sc1;
            if (k0 < LC) {
                float bbv = btab[hT + rpi[ridx + k0]];
                sc0 = s0[r] + (maskp[mrow + k0] ? bbv : NEG_INF);
            } else sc0 = NEG_INF;
            if (k1 < LC) {
                float bbv = btab[hT + rpi[ridx + k1]];
                sc1 = s1[r] + (maskp[mrow + k1] ? bbv : NEG_INF);
            } else sc1 = NEG_INF;

            float mb2 = redmax16(fmaxf(sc0, sc1));
            float mn  = fmaxf(m[r], mb2);
            float p0 = __expf(sc0 - mn);
            float p1 = __expf(sc1 - mn);
            float alpha = __expf(m[r] - mn);
            float rs = redsum16(p0 + p1);
            s[r] = s[r] * alpha + rs;
            m[r] = mn;
            o0[r] *= alpha; o1[r] *= alpha;

            const int r0 = r + 8 * half;
            Pw[r0 * LDA + nl]      = (__bf16)p0;
            Pw[r0 * LDA + 16 + nl] = (__bf16)p1;
        }

        // ---- PV: P(16x32) @ V(32x32), CH split into two N=16 halves
        v16bf ap  = ld_frag_a(Pw, LDA);
        v16bf bv0 = ld_frag_b(sV, LDA);
        v16bf bv1 = ld_frag_b(sV + 16 * LDA, LDA);
        o0 = wmma_bf16(ap, bv0, o0);
        o1 = wmma_bf16(ap, bv1, o1);
        __syncthreads();
    }

    // ---- normalize + store to attn_out [row = n*L+l][DIM] bf16
#pragma unroll
    for (int r = 0; r < 8; ++r) {
        const int qg = qbase + r + 8 * half;
        if (qg < LC) {
            const float inv = 1.0f / s[r];
            const size_t rowb = ((size_t)n * LC + qg) * DIMC + h * CHC;
            abuf[rowb + nl]      = (__bf16)(o0[r] * inv);
            abuf[rowb + 16 + nl] = (__bf16)(o1[r] * inv);
        }
    }
}

// ---------------- kernel 3: output projection + bias -----------------------
// grid (8, 343): x = N-tile over 512, y = M-tile over 21952.

__global__ void __launch_bounds__(128)
proj_gemm(const __bf16* __restrict__ abf, const __bf16* __restrict__ wp,
          const float* __restrict__ bias, float* __restrict__ out) {
    __shared__ __bf16 sA[2 * BUFE];
    __shared__ __bf16 sB[2 * BUFE];
    const int mb = blockIdx.y * 64;
    const int nb = blockIdx.x * 64;
    v8f acc[4] = {{0,0,0,0,0,0,0,0},{0,0,0,0,0,0,0,0},{0,0,0,0,0,0,0,0},{0,0,0,0,0,0,0,0}};
    gemm_mainloop(abf, wp, mb, nb, sA, sB, acc);

    const int lane = threadIdx.x & 31, half = lane >> 4, nl = lane & 15;
    const int w = threadIdx.x >> 5;
#pragma unroll
    for (int t = 0; t < 4; ++t) {
        const int gcol = nb + t * 16 + nl;
        const float bv = bias[gcol];
#pragma unroll
        for (int r = 0; r < 8; ++r) {
            const int grow = mb + w * 16 + r + 8 * half;
            out[(size_t)grow * DIMC + gcol] = acc[t][r] + bv;
        }
    }
}

// ---------------------------------------------------------------------------

extern "C" void kernel_launch(void* const* d_in, const int* in_sizes, int n_in,
                              void* d_out, int out_size, void* d_ws, size_t ws_size,
                              hipStream_t stream) {
    (void)in_sizes; (void)n_in; (void)out_size; (void)ws_size;

    const float*         x     = (const float*)d_in[0];
    const unsigned char* mask  = (const unsigned char*)d_in[1]; // numpy bool = 1 byte
    const int*           rpi   = (const int*)d_in[2];
    const float*         Wqkv  = (const float*)d_in[3];
    const float*         Wproj = (const float*)d_in[4];
    const float*         bproj = (const float*)d_in[5];
    const float*         btab  = (const float*)d_in[6];
    float*               out   = (float*)d_out;

    // workspace carve-up (all 256B aligned)
    constexpr size_t XBF_B  = (size_t)ROWS * DIMC * 2;           // 22,478,848
    constexpr size_t WQ_B   = (size_t)3 * DIMC * DIMC * 2;       //  1,572,864
    constexpr size_t WP_B   = (size_t)DIMC * DIMC * 2;           //    524,288
    constexpr size_t HEAD_B = XBF_B;                              // q/k/v each
    char* ws = (char*)d_ws;
    __bf16* xbf   = (__bf16*)ws;                 ws += XBF_B;
    __bf16* wqkvb = (__bf16*)ws;                 ws += WQ_B;
    __bf16* wprojb= (__bf16*)ws;                 ws += WP_B;
    __bf16* qb    = (__bf16*)ws;                 ws += HEAD_B;
    __bf16* kbuf  = (__bf16*)ws;                 ws += HEAD_B;
    __bf16* vbuf  = (__bf16*)ws;                 ws += HEAD_B;
    __bf16* abuf  = (__bf16*)ws;                 ws += HEAD_B;

    // 1) fp32 -> bf16 conversions
    cvt_bf16<<<4096, 256, 0, stream>>>(x, xbf, ROWS * DIMC);
    cvt_bf16<<<1024, 256, 0, stream>>>(Wqkv, wqkvb, 3 * DIMC * DIMC);
    cvt_bf16<<<512, 256, 0, stream>>>(Wproj, wprojb, DIMC * DIMC);

    // 2) QKV projection (scatter into per-head q/k/v, q pre-scaled)
    qkv_gemm<<<dim3(24, 343), 128, 0, stream>>>(xbf, wqkvb, qb, kbuf, vbuf);

    // 3) fused flash attention
    attn_kernel<<<dim3(6, NHC, NWIN), 128, 0, stream>>>(qb, kbuf, vbuf, rpi, btab, mask, abuf);

    // 4) output projection + bias -> fp32
    proj_gemm<<<dim3(8, 343), 128, 0, stream>>>(abuf, wprojb, bproj, out);
}
